// CausalSelfAttention_88819923682027
// MI455X (gfx1250) — matrix operand order repaired and checked by reading, not compiled, not run on hardware
//
#include <hip/hip_runtime.h>
#include <hip/hip_bf16.h>

// ---------------------------------------------------------------------------
// Causal self-attention block for MI455X (gfx1250), bf16 WMMA + fp32 accum.
// T=2048, D=2048, H=16, C=128.
// GEMMs double-buffered via GLOBAL_LOAD_ASYNC_TO_LDS_B128 when available.
// ---------------------------------------------------------------------------

typedef __attribute__((ext_vector_type(16))) __bf16 v16bf;
typedef __attribute__((ext_vector_type(8)))  float  v8f;

union BFrag {
  v16bf v;
  uint4 q[2];
};
static_assert(sizeof(BFrag) == 32, "frag size");

__device__ __forceinline__ v8f wmma_bf16(const BFrag& a, const BFrag& b, v8f c) {
  // D = A(16x32) * B(32x16) + C, fp32 accumulate
  return __builtin_amdgcn_wmma_f32_16x16x32_bf16(
      /*neg_a=*/false, a.v, /*neg_b=*/false, b.v,
      /*c_mod=*/(short)0, c, /*reuse_a=*/false, /*reuse_b=*/false);
}

__device__ __forceinline__ v8f zero_v8f() { v8f z = {}; return z; }

constexpr int T  = 2048;
constexpr int D  = 2048;
constexpr int H  = 16;
constexpr int Cd = 128;        // head dim
constexpr int threeD = 3 * D;

// ---- async global->LDS copy path (guarded; falls back to VGPR staging) ----
#if defined(__gfx1250__) && __has_builtin(__builtin_amdgcn_global_load_async_to_lds_b128)
#define HAVE_ASYNC_LDS 1
#else
#define HAVE_ASYNC_LDS 0
#endif

#if HAVE_ASYNC_LDS
typedef int v4i_gcc __attribute__((vector_size(16)));
typedef __attribute__((address_space(1))) v4i_gcc* gptr_b128;
typedef __attribute__((address_space(3))) v4i_gcc* lptr_b128;

__device__ __forceinline__ void async_b128(const void* g, void* l) {
  __builtin_amdgcn_global_load_async_to_lds_b128(
      (gptr_b128)g, (lptr_b128)l, /*offset=*/0, /*cpol=*/0);
}
__device__ __forceinline__ void wait_async() {
#if __has_builtin(__builtin_amdgcn_s_wait_asynccnt)
  __builtin_amdgcn_s_wait_asynccnt(0);
#else
  asm volatile("s_wait_asynccnt 0x0" ::: "memory");
#endif
}
#else
__device__ __forceinline__ void wait_async() {}
#endif

// ---------------------------------------------------------------------------
// fp32 -> bf16 conversion (vectorized by 4; n must be a multiple of 1024)
// ---------------------------------------------------------------------------
__global__ __launch_bounds__(256)
void f32_to_bf16_kernel(const float* __restrict__ in,
                        __hip_bfloat16* __restrict__ out, int n) {
  int idx = (blockIdx.x * 256 + threadIdx.x) * 4;
  if (idx + 3 < n) {
    float4 f = *reinterpret_cast<const float4*>(in + idx);
    out[idx + 0] = __float2bfloat16(f.x);
    out[idx + 1] = __float2bfloat16(f.y);
    out[idx + 2] = __float2bfloat16(f.z);
    out[idx + 3] = __float2bfloat16(f.w);
  }
}

// ---------------------------------------------------------------------------
// Generic Y[M,N] = A[M,K] @ B[N,K]^T   (i.e. y = a @ w^T, weights row-major)
// Block tile 128x128, K-step 32, 256 threads = 8 waves (4x2 wave grid),
// each wave computes a 32x64 tile = 8 WMMA accumulators.
// Double-buffered LDS; next K-slab staged asynchronously while WMMAs run.
// ---------------------------------------------------------------------------
template <bool OUT_BF16>
__global__ __launch_bounds__(256)
void gemm_xwT(const __hip_bfloat16* __restrict__ A,
              const __hip_bfloat16* __restrict__ B,
              void* __restrict__ Cp, int M, int N, int K) {
  constexpr int BM = 128, BN = 128, BK = 32, PAD = 8;
  __shared__ alignas(16) __hip_bfloat16 sA[2][BM][BK + PAD];
  __shared__ alignas(16) __hip_bfloat16 sB[2][BN][BK + PAD];

  const int tid  = threadIdx.x;
  const int lane = tid & 31;
  const int wv   = tid >> 5;      // wave 0..7
  const int hx   = lane >> 4;     // lane half (0/1)
  const int r    = lane & 15;

  const int m0 = blockIdx.y * BM;
  const int n0 = blockIdx.x * BN;
  const int wm = (wv >> 1) * 32;  // wave row offset: 0,32,64,96
  const int wn = (wv & 1) * 64;   // wave col offset: 0,64

  v8f acc[2][4];
#pragma unroll
  for (int i = 0; i < 2; ++i)
#pragma unroll
    for (int j = 0; j < 4; ++j) acc[i][j] = zero_v8f();

  const int ldRow = tid >> 1;        // 0..127
  const int ldSeg = (tid & 1) * 16;  // element offset 0/16

  auto stage = [&](int buf, int k0) {
    const __hip_bfloat16* ga = A + (size_t)(m0 + ldRow) * K + k0 + ldSeg;
    const __hip_bfloat16* gb = B + (size_t)(n0 + ldRow) * K + k0 + ldSeg;
#if HAVE_ASYNC_LDS
    async_b128(ga,     &sA[buf][ldRow][ldSeg]);
    async_b128(ga + 8, &sA[buf][ldRow][ldSeg + 8]);
    async_b128(gb,     &sB[buf][ldRow][ldSeg]);
    async_b128(gb + 8, &sB[buf][ldRow][ldSeg + 8]);
#else
    const uint4* pa = reinterpret_cast<const uint4*>(ga);
    const uint4* pb = reinterpret_cast<const uint4*>(gb);
    uint4 a0 = pa[0], a1 = pa[1], b0 = pb[0], b1 = pb[1];
    *reinterpret_cast<uint4*>(&sA[buf][ldRow][ldSeg])     = a0;
    *reinterpret_cast<uint4*>(&sA[buf][ldRow][ldSeg + 8]) = a1;
    *reinterpret_cast<uint4*>(&sB[buf][ldRow][ldSeg])     = b0;
    *reinterpret_cast<uint4*>(&sB[buf][ldRow][ldSeg + 8]) = b1;
#endif
  };

  stage(0, 0);
  wait_async();
  __syncthreads();

  const int NK = K / BK;
  for (int kb = 0; kb < NK; ++kb) {
    const int cur = kb & 1;
    if (kb + 1 < NK) stage(cur ^ 1, (kb + 1) * BK);  // prefetch next slab

    // --- fragments (A: row r, chunks hx*8 and 16+hx*8; B: same chunks) ---
    BFrag af[2], bfr[4];
#pragma unroll
    for (int i = 0; i < 2; ++i) {
      const __hip_bfloat16* p = &sA[cur][wm + i * 16 + r][0];
      af[i].q[0] = *reinterpret_cast<const uint4*>(p + hx * 8);
      af[i].q[1] = *reinterpret_cast<const uint4*>(p + 16 + hx * 8);
    }
#pragma unroll
    for (int j = 0; j < 4; ++j) {
      const __hip_bfloat16* p = &sB[cur][wn + j * 16 + r][0];
      bfr[j].q[0] = *reinterpret_cast<const uint4*>(p + hx * 8);
      bfr[j].q[1] = *reinterpret_cast<const uint4*>(p + 16 + hx * 8);
    }
#pragma unroll
    for (int i = 0; i < 2; ++i)
#pragma unroll
      for (int j = 0; j < 4; ++j) acc[i][j] = wmma_bf16(af[i], bfr[j], acc[i][j]);

    wait_async();     // this wave's prefetch into buf^1 has landed
    __syncthreads();  // all waves done reading cur / writing cur^1
  }

  // --- epilogue: C/D layout -> row = hx*8 + e, col = r ---
#pragma unroll
  for (int i = 0; i < 2; ++i)
#pragma unroll
    for (int j = 0; j < 4; ++j)
#pragma unroll
      for (int e = 0; e < 8; ++e) {
        int row = m0 + wm + i * 16 + hx * 8 + e;
        int col = n0 + wn + j * 16 + r;
        float val = acc[i][j][e];
        if (OUT_BF16)
          ((__hip_bfloat16*)Cp)[(size_t)row * N + col] = __float2bfloat16(val);
        else
          ((float*)Cp)[(size_t)row * N + col] = val;
      }
}

// ---------------------------------------------------------------------------
// Flash attention: grid = (T/64, H), block = 128 threads = 4 waves.
// Wave w handles 16 query rows; WG shares K-tile [64][128] and V^T [128][64].
// ---------------------------------------------------------------------------
__global__ __launch_bounds__(128)
void attn_kernel(const __hip_bfloat16* __restrict__ qkv,  // [T][3D]
                 __hip_bfloat16* __restrict__ ao) {       // [T][D]
  constexpr int PADK = 8;
  __shared__ alignas(16) __hip_bfloat16 sK[64][Cd + PADK];      // [s][c]
  __shared__ alignas(16) __hip_bfloat16 sVT[Cd][64 + PADK];     // [c][s]
  __shared__ alignas(16) __hip_bfloat16 sP[4][16][64 + PADK];   // per-wave P

  const int tid  = threadIdx.x;
  const int lane = tid & 31;
  const int wv   = tid >> 5;   // 0..3
  const int hx   = lane >> 4;
  const int r    = lane & 15;

  const int qb = blockIdx.x;       // 64-row query block
  const int h  = blockIdx.y;
  const int qw = qb * 64 + wv * 16;

  const float scale = 0.08838834764831845f;  // 1/sqrt(128)

  // Q fragments (A layout), straight from global, 4 K-steps over C=128
  BFrag qf[4];
  {
    const __hip_bfloat16* qrow = qkv + (size_t)(qw + r) * threeD + h * Cd;
#pragma unroll
    for (int ks = 0; ks < 4; ++ks) {
      qf[ks].q[0] = *reinterpret_cast<const uint4*>(qrow + ks * 32 + hx * 8);
      qf[ks].q[1] = *reinterpret_cast<const uint4*>(qrow + ks * 32 + 16 + hx * 8);
    }
  }

  v8f o[8];
#pragma unroll
  for (int t = 0; t < 8; ++t) o[t] = zero_v8f();
  float mrow[8], lrow[8];
#pragma unroll
  for (int j = 0; j < 8; ++j) { mrow[j] = -__builtin_inff(); lrow[j] = 0.0f; }

  for (int kb = 0; kb <= qb; ++kb) {
    const int s0 = kb * 64;
    __syncthreads();  // previous tiles fully consumed

    // stage K tile: 64x128 bf16, 128B per thread (async LDS copy if available)
    {
      int rowK = tid >> 1;
      int segK = (tid & 1) * 64;
      const __hip_bfloat16* g =
          qkv + (size_t)(s0 + rowK) * threeD + D + h * Cd + segK;
#if HAVE_ASYNC_LDS
#pragma unroll
      for (int i = 0; i < 8; ++i)
        async_b128(g + i * 8, &sK[rowK][segK + i * 8]);
#else
      const uint4* gq = reinterpret_cast<const uint4*>(g);
      uint4* dst = reinterpret_cast<uint4*>(&sK[rowK][segK]);
#pragma unroll
      for (int i = 0; i < 8; ++i) dst[i] = gq[i];
#endif
    }
    // stage V transposed: read 8 bf16 along c, scatter into sVT[c][s]
    {
#pragma unroll
      for (int i = 0; i < 8; ++i) {
        int id = tid * 8 + i;     // 0..1023
        int sv = id >> 4;         // key row 0..63
        int c0 = (id & 15) * 8;   // channel chunk
        uint4 d = *reinterpret_cast<const uint4*>(
            qkv + (size_t)(s0 + sv) * threeD + 2 * D + h * Cd + c0);
        const __hip_bfloat16* e = reinterpret_cast<const __hip_bfloat16*>(&d);
#pragma unroll
        for (int j = 0; j < 8; ++j) sVT[c0 + j][sv] = e[j];
      }
    }
    wait_async();
    __syncthreads();

    // ---- scores S (16x64) = Q K^T : 4 n-tiles x 4 K-steps = 16 WMMA ----
    v8f s[4];
#pragma unroll
    for (int n = 0; n < 4; ++n) s[n] = zero_v8f();
#pragma unroll
    for (int n = 0; n < 4; ++n)
#pragma unroll
      for (int ks = 0; ks < 4; ++ks) {
        BFrag kf;
        const __hip_bfloat16* p = &sK[n * 16 + r][0];
        kf.q[0] = *reinterpret_cast<const uint4*>(p + ks * 32 + hx * 8);
        kf.q[1] = *reinterpret_cast<const uint4*>(p + ks * 32 + 16 + hx * 8);
        s[n] = wmma_bf16(qf[ks], kf, s[n]);
      }

    // ---- scale + causal mask (only the diagonal block needs it) ----
#pragma unroll
    for (int n = 0; n < 4; ++n)
#pragma unroll
      for (int j = 0; j < 8; ++j) {
        float v = s[n][j] * scale;
        if (kb == qb) {
          int qrow = qw + hx * 8 + j;
          int kcol = s0 + n * 16 + r;
          v = (kcol > qrow) ? -__builtin_inff() : v;
        }
        s[n][j] = v;
      }

    // ---- online softmax (row = hx*8 + j; reduce across 16 lanes/half) ----
    float mnew[8], alpha[8];
#pragma unroll
    for (int j = 0; j < 8; ++j) {
      float mx = fmaxf(fmaxf(s[0][j], s[1][j]), fmaxf(s[2][j], s[3][j]));
#pragma unroll
      for (int off = 8; off >= 1; off >>= 1) mx = fmaxf(mx, __shfl_xor(mx, off, 32));
      mnew[j]  = fmaxf(mrow[j], mx);
      alpha[j] = __expf(mrow[j] - mnew[j]);
      mrow[j]  = mnew[j];
    }
#pragma unroll
    for (int n = 0; n < 4; ++n)
#pragma unroll
      for (int j = 0; j < 8; ++j) s[n][j] = __expf(s[n][j] - mnew[j]);
#pragma unroll
    for (int j = 0; j < 8; ++j) {
      float sm = s[0][j] + s[1][j] + s[2][j] + s[3][j];
#pragma unroll
      for (int off = 8; off >= 1; off >>= 1) sm += __shfl_xor(sm, off, 32);
      lrow[j] = lrow[j] * alpha[j] + sm;
    }
#pragma unroll
    for (int t = 0; t < 8; ++t)
#pragma unroll
      for (int j = 0; j < 8; ++j) o[t][j] *= alpha[j];

    // ---- P: C/D layout -> LDS -> A layout ----
#pragma unroll
    for (int n = 0; n < 4; ++n)
#pragma unroll
      for (int j = 0; j < 8; ++j)
        sP[wv][hx * 8 + j][n * 16 + r] = __float2bfloat16(s[n][j]);
    __syncthreads();

    BFrag pf[2];
#pragma unroll
    for (int ks = 0; ks < 2; ++ks) {
      const __hip_bfloat16* p = &sP[wv][r][0];
      pf[ks].q[0] = *reinterpret_cast<const uint4*>(p + ks * 32 + hx * 8);
      pf[ks].q[1] = *reinterpret_cast<const uint4*>(p + ks * 32 + 16 + hx * 8);
    }

    // ---- O += P V : 8 channel tiles x 2 K-steps = 16 WMMA ----
#pragma unroll
    for (int ct = 0; ct < 8; ++ct)
#pragma unroll
      for (int ks = 0; ks < 2; ++ks) {
        BFrag vf;
        const __hip_bfloat16* p = &sVT[ct * 16 + r][0];
        vf.q[0] = *reinterpret_cast<const uint4*>(p + ks * 32 + hx * 8);
        vf.q[1] = *reinterpret_cast<const uint4*>(p + ks * 32 + 16 + hx * 8);
        o[ct] = wmma_bf16(pf[ks], vf, o[ct]);
      }
  }

  // ---- epilogue: normalize and store bf16 [T][D] ----
#pragma unroll
  for (int ct = 0; ct < 8; ++ct)
#pragma unroll
    for (int j = 0; j < 8; ++j) {
      int row = qw + hx * 8 + j;
      int col = h * Cd + ct * 16 + r;
      ao[(size_t)row * D + col] = __float2bfloat16(o[ct][j] / lrow[j]);
    }
}

// ---------------------------------------------------------------------------
extern "C" void kernel_launch(void* const* d_in, const int* in_sizes, int n_in,
                              void* d_out, int out_size, void* d_ws, size_t ws_size,
                              hipStream_t stream) {
  (void)in_sizes; (void)n_in; (void)out_size; (void)ws_size;

  const float* x     = (const float*)d_in[0];  // [T, D]
  const float* Wattn = (const float*)d_in[1];  // [3D, D]
  const float* Wproj = (const float*)d_in[2];  // [D, D]
  float* out = (float*)d_out;                  // [T, D] fp32

  char* ws = (char*)d_ws;
  __hip_bfloat16* x_bf   = (__hip_bfloat16*)ws; ws += (size_t)T * D * 2;
  __hip_bfloat16* Wa_bf  = (__hip_bfloat16*)ws; ws += (size_t)3 * D * D * 2;
  __hip_bfloat16* Wp_bf  = (__hip_bfloat16*)ws; ws += (size_t)D * D * 2;
  __hip_bfloat16* qkv_bf = (__hip_bfloat16*)ws; ws += (size_t)T * 3 * D * 2;
  __hip_bfloat16* ao_bf  = (__hip_bfloat16*)ws;

  f32_to_bf16_kernel<<<(T * D) / 1024, 256, 0, stream>>>(x, x_bf, T * D);
  f32_to_bf16_kernel<<<(3 * D * D) / 1024, 256, 0, stream>>>(Wattn, Wa_bf, 3 * D * D);
  f32_to_bf16_kernel<<<(D * D) / 1024, 256, 0, stream>>>(Wproj, Wp_bf, D * D);

  // qkv = x @ W_attn^T   (bf16 out)
  gemm_xwT<true><<<dim3((3 * D) / 128, T / 128), 256, 0, stream>>>(
      x_bf, Wa_bf, (void*)qkv_bf, T, 3 * D, D);

  // flash attention -> ao (bf16 [T][D])
  attn_kernel<<<dim3(T / 64, H), 128, 0, stream>>>(qkv_bf, ao_bf);

  // out = ao @ W_proj^T  (fp32 out)
  gemm_xwT<false><<<dim3(D / 128, T / 128), 256, 0, stream>>>(
      ao_bf, Wp_bf, (void*)out, T, D, D);
}
